// MultiHeadSWCA_45251775430662
// MI455X (gfx1250) — matrix operand order repaired and checked by reading, not compiled, not run on hardware
//
#include <hip/hip_runtime.h>
#include <hip/hip_bf16.h>

// ---------------------------------------------------------------------------
// Shifted-window multi-head cross attention for MI455X (gfx1250, wave32).
// One workgroup (128 threads = 4 waves) per (batch, head, window) = 6144 WGs.
// All six 64x64x64 GEMMs per window run on v_wmma_f32_16x16x32_f16 with f32
// accumulation, entirely out of LDS. Each wave owns a 16-row strip.
// ---------------------------------------------------------------------------

#define WINDOW   8
#define HEADS    12
#define HEAD_DIM 64
#define NWIN     64          // 8x8 windows in a 64x64 image
#define BATCH    8
#define IMG      64
#define CTOT     (HEADS * HEAD_DIM)   // 768
#define LDH      72          // f16 row stride for 64-wide LDS tiles (144B, 16B-aligned rows)
#define SLD      68          // f32 row stride for the score tile

typedef __attribute__((ext_vector_type(8)))  _Float16 v8h;
typedef __attribute__((ext_vector_type(16))) _Float16 v16h;
typedef __attribute__((ext_vector_type(8)))  float    v8f;

// ---- LDS layout (bytes); phases overlay dead buffers -----------------------
#define TILE_B    (64 * LDH * 2)           // 9216 B per 64x64 f16 tile
#define OFF_WQ    0
#define OFF_WK    (OFF_WQ + TILE_B)
#define OFF_WV    (OFF_WK + TILE_B)
#define OFF_WL    (OFF_WV + TILE_B)
#define OFF_RAWQ  (OFF_WL + TILE_B)        // 36864
#define OFF_RAWK  (OFF_RAWQ + TILE_B)
#define OFF_RAWV  (OFF_RAWK + TILE_B)      // raw slab ends 64512
#define OFF_SCORE OFF_RAWQ                 // f32 64xSLD, overlays raw slab (dead)
#define OFF_ATT16 (OFF_SCORE + 64 * SLD * 4)   // 54272
#define OFF_STATS (OFF_ATT16 + TILE_B)     // 63488: rowmax[64], rowrinv[64]
#define OFF_QP    (OFF_RAWV + TILE_B)      // 64512
#define OFF_KP    (OFF_QP + TILE_B)
#define OFF_VT    (OFF_KP + TILE_B)
#define OFF_O16   OFF_QP                   // overlays Qp (dead after scores)
#define OFF_PE    (OFF_VT + TILE_B)        // 92160: 15x15 f32 bias table
#define SMEM_BYTES (OFF_PE + 15 * 15 * 4 + 64)

union V16U { v16h v; v8h h[2]; };

// A fragment (16x32 f16): lane -> M=lane%16;
// VGPR0-7 hold K = kb..kb+15, kb = lane<16 ? 0 : 16 (matches B fragment).
__device__ __forceinline__ v16h ld_afrag(const _Float16* A, int m0, int ks, int lane) {
  int m  = m0 + (lane & 15);
  int kb = ks * 32 + ((lane & 16) ? 16 : 0);
  const _Float16* p = A + m * LDH + kb;
  V16U u;
  u.h[0] = *(const v8h*)(p);        // K = kb .. kb+7
  u.h[1] = *(const v8h*)(p + 8);    // K = kb+8 .. kb+15
  return u.v;
}

// B fragment (32x16 f16): lane -> N=lane%16; VGPRj holds K = kb+2j, kb+2j+1,
// kb = lane<16 ? 0 : 16.  Brm is the row-major TRANSPOSE of B: Brm[N][K],
// so each lane reads 16 consecutive halves (32 contiguous bytes).
__device__ __forceinline__ v16h ld_bfrag(const _Float16* Brm, int nt, int ks, int lane) {
  int n  = nt * 16 + (lane & 15);
  int kb = ks * 32 + ((lane & 16) ? 16 : 0);
  const _Float16* p = Brm + n * LDH + kb;
  V16U u;
  u.h[0] = *(const v8h*)(p);
  u.h[1] = *(const v8h*)(p + 8);
  return u.v;
}

// One 16x64 strip of a 64x64x64 GEMM: D(strip) = A(strip, 64) * Brm^T(64,64).
__device__ __forceinline__ void gemm64(const _Float16* A, const _Float16* Brm,
                                       int m0, int lane, v8f acc[4]) {
#pragma unroll
  for (int ks = 0; ks < 2; ++ks) {
    v16h a = ld_afrag(A, m0, ks, lane);
#pragma unroll
    for (int nt = 0; nt < 4; ++nt) {
      v16h b = ld_bfrag(Brm, nt, ks, lane);
      acc[nt] = __builtin_amdgcn_wmma_f32_16x16x32_f16(
          /*neg_a=*/false, a, /*neg_b=*/false, b,
          /*c_mod=*/(short)0, acc[nt], /*reuse_a=*/false, /*reuse_b=*/false);
    }
  }
}

__global__ __launch_bounds__(128) void swca_kernel(
    const float* __restrict__ q_feat, const float* __restrict__ k_feat,
    const float* __restrict__ v_feat, const float* __restrict__ Wq,
    const float* __restrict__ Wk, const float* __restrict__ Wv,
    const float* __restrict__ Wlin, const float* __restrict__ pe,
    float* __restrict__ out, float* __restrict__ attn_out) {
  __shared__ __align__(16) unsigned char smem[SMEM_BYTES];
  _Float16* wqL   = (_Float16*)(smem + OFF_WQ);
  _Float16* wkL   = (_Float16*)(smem + OFF_WK);
  _Float16* wvL   = (_Float16*)(smem + OFF_WV);
  _Float16* wlL   = (_Float16*)(smem + OFF_WL);
  _Float16* rawQ  = (_Float16*)(smem + OFF_RAWQ);
  _Float16* rawK  = (_Float16*)(smem + OFF_RAWK);
  _Float16* rawV  = (_Float16*)(smem + OFF_RAWV);
  float*    score = (float*)   (smem + OFF_SCORE);
  _Float16* att16 = (_Float16*)(smem + OFF_ATT16);
  float*    rowmx = (float*)   (smem + OFF_STATS);
  float*    rowri = (float*)   (smem + OFF_STATS) + 64;
  _Float16* Qp    = (_Float16*)(smem + OFF_QP);
  _Float16* Kp    = (_Float16*)(smem + OFF_KP);
  _Float16* Vt    = (_Float16*)(smem + OFF_VT);
  _Float16* O16   = (_Float16*)(smem + OFF_O16);
  float*    peL   = (float*)   (smem + OFF_PE);

  const int tid  = threadIdx.x;
  const int lane = tid & 31;
  const int wave = tid >> 5;
  const int bid  = blockIdx.x;
  const int n    = bid & 63;               // window index
  const int hh   = (bid >> 6) % HEADS;     // head
  const int bb   = bid / (64 * HEADS);     // batch
  const int wy   = n >> 3, wx = n & 7;

  // ---- Phase 1: gather rolled window tiles + weights into LDS (f32 -> f16) --
  {
    int s = tid >> 1, d0 = (tid & 1) * 32;     // 2 threads per token row
    int ry = s >> 3, rx = s & 7;
    int y = (wy * 8 + ry + 4) & 63;            // roll(-DISP) gather: src = (pos + 4) mod 64
    int x = (wx * 8 + rx + 4) & 63;
    long base = ((long)bb * 4096 + y * 64 + x) * CTOT + hh * HEAD_DIM + d0;
    const float* qs = q_feat + base;
    const float* ks = k_feat + base;
    const float* vs = v_feat + base;
    _Float16* dq = rawQ + s * LDH + d0;
    _Float16* dk = rawK + s * LDH + d0;
    _Float16* dv = rawV + s * LDH + d0;
#pragma unroll 8
    for (int d = 0; d < 32; ++d) {
      dq[d] = (_Float16)qs[d];
      dk[d] = (_Float16)ks[d];
      dv[d] = (_Float16)vs[d];
    }
  }
  for (int i = tid; i < 4096; i += 128) {      // W stored row-major [e][d] == B^T
    int r = i >> 6, c = i & 63;
    wqL[r * LDH + c] = (_Float16)Wq[i];
    wkL[r * LDH + c] = (_Float16)Wk[i];
    wvL[r * LDH + c] = (_Float16)Wv[i];
    wlL[r * LDH + c] = (_Float16)Wlin[i];
  }
  for (int i = tid; i < 225; i += 128) peL[i] = pe[i];
  __syncthreads();

  const int m0     = wave * 16;
  const int halfid = lane >> 4;
  const int nl     = lane & 15;

  // ---- Phase 2: Q/K/V projections (X @ W^T), store Q,K row-major, V transposed
  {
    v8f aq[4] = {}; gemm64(rawQ, wqL, m0, lane, aq);
    v8f ak[4] = {}; gemm64(rawK, wkL, m0, lane, ak);
    v8f av[4] = {}; gemm64(rawV, wvL, m0, lane, av);
#pragma unroll
    for (int nt = 0; nt < 4; ++nt)
#pragma unroll
      for (int r = 0; r < 8; ++r) {
        int mm = m0 + halfid * 8 + r;          // token
        int nn = nt * 16 + nl;                 // dim e
        Qp[mm * LDH + nn] = (_Float16)aq[nt][r];
        Kp[mm * LDH + nn] = (_Float16)ak[nt][r];
        Vt[nn * LDH + mm] = (_Float16)av[nt][r];   // transposed: [d][token]
      }
  }
  __syncthreads();

  // ---- Phase 3: S = Q K^T / 8 + PE bias + shift-window mask -----------------
  {
    v8f sc[4] = {}; gemm64(Qp, Kp, m0, lane, sc);
    const bool mUL = (wy == 7), mLR = (wx == 7);
#pragma unroll
    for (int nt = 0; nt < 4; ++nt)
#pragma unroll
      for (int r = 0; r < 8; ++r) {
        int q = m0 + halfid * 8 + r;
        int k = nt * 16 + nl;
        float s = sc[nt][r] * 0.125f;
        s += peL[((k >> 3) - (q >> 3) + 7) * 15 + ((k & 7) - (q & 7) + 7)];
        if (mUL && ((q >= 32) != (k >= 32)))           s = -__builtin_inff();
        if (mLR && (((q & 7) >= 4) != ((k & 7) >= 4))) s = -__builtin_inff();
        score[q * SLD + k] = s;
      }
  }
  __syncthreads();

  // ---- Phase 3b: row softmax stats, then write attn (global f32 + LDS f16) --
  if (tid < 64) {
    const float* row = score + tid * SLD;
    float mx = -__builtin_inff();
    for (int k = 0; k < 64; ++k) mx = fmaxf(mx, row[k]);
    float sum = 0.f;
    for (int k = 0; k < 64; ++k) sum += __expf(row[k] - mx);
    rowmx[tid] = mx;
    rowri[tid] = 1.0f / sum;
  }
  __syncthreads();
  {
    int q = tid >> 1, c0 = (tid & 1) * 32;
    float mx = rowmx[q], ri = rowri[q];
    const float* srow = score + q * SLD + c0;
    float* ga = attn_out + ((((long)bb * HEADS + hh) * NWIN + n) * 64 + q) * 64 + c0;
    _Float16* arow = att16 + q * LDH + c0;
#pragma unroll 8
    for (int k = 0; k < 32; ++k) {
      float p = __expf(srow[k] - mx) * ri;
      ga[k]   = p;
      arow[k] = (_Float16)p;
    }
  }
  __syncthreads();

  // ---- Phase 4: O = attn @ V  (B^T supplied by transposed Vt) ---------------
  {
    v8f ov[4] = {}; gemm64(att16, Vt, m0, lane, ov);
#pragma unroll
    for (int nt = 0; nt < 4; ++nt)
#pragma unroll
      for (int r = 0; r < 8; ++r)
        O16[(m0 + halfid * 8 + r) * LDH + nt * 16 + nl] = (_Float16)ov[nt][r];
  }
  __syncthreads();

  // ---- Phase 5: out = O @ Wlin^T, scatter with inverse roll/window map ------
  {
    v8f oo[4] = {}; gemm64(O16, wlL, m0, lane, oo);
#pragma unroll
    for (int nt = 0; nt < 4; ++nt)
#pragma unroll
      for (int r = 0; r < 8; ++r) {
        int s  = m0 + halfid * 8 + r;
        int e  = nt * 16 + nl;
        int ry = s >> 3, rx = s & 7;
        int y  = (wy * 8 + ry + 4) & 63;       // inverse map == forward map
        int x  = (wx * 8 + rx + 4) & 63;
        out[((long)bb * 4096 + y * 64 + x) * CTOT + hh * HEAD_DIM + e] = oo[nt][r];
      }
  }
}

extern "C" void kernel_launch(void* const* d_in, const int* in_sizes, int n_in,
                              void* d_out, int out_size, void* d_ws, size_t ws_size,
                              hipStream_t stream) {
  (void)in_sizes; (void)n_in; (void)d_ws; (void)ws_size; (void)out_size;
  const float* q_feat = (const float*)d_in[0];
  const float* k_feat = (const float*)d_in[1];
  const float* v_feat = (const float*)d_in[2];
  const float* Wq     = (const float*)d_in[3];
  const float* Wk     = (const float*)d_in[4];
  const float* Wv     = (const float*)d_in[5];
  const float* Wlin   = (const float*)d_in[6];
  const float* pe     = (const float*)d_in[7];
  float* out  = (float*)d_out;
  float* attn = out + (size_t)BATCH * 4096 * CTOT;   // tuple: (out, attn) concatenated

  dim3 grid(BATCH * HEADS * NWIN);   // 6144 workgroups
  dim3 block(128);                   // 4 waves (wave32)
  hipLaunchKernelGGL(swca_kernel, grid, block, 0, stream,
                     q_feat, k_feat, v_feat, Wq, Wk, Wv, Wlin, pe, out, attn);
}
